// CompGCNCov_74414603370818
// MI455X (gfx1250) — matrix operand order repaired and checked by reading, not compiled, not run on hardware
//
#include <hip/hip_runtime.h>
#include <hip/hip_bf16.h>

typedef __attribute__((ext_vector_type(16))) _Float16 v16h;
typedef __attribute__((ext_vector_type(8)))  float    v8f;

#define V_NODES 100000
#define E_EDGES 300000
#define IN_CH   256
#define OUT_CH  256
#define NREL2   474
#define KDIM    768            // 512 (two directions) + 256 (self loop)
#define BN_EPS  1e-5f

// ---------------------------------------------------------------------------
// Utility: zero a float region (grid-stride)
// ---------------------------------------------------------------------------
__global__ void k_zero(float* __restrict__ p, long long n) {
  long long i = (long long)blockIdx.x * blockDim.x + threadIdx.x;
  long long stride = (long long)gridDim.x * blockDim.x;
  for (; i < n; i += stride) p[i] = 0.0f;
}

// ---------------------------------------------------------------------------
// In-degree via float atomics (counts are small integers; exact in fp32)
// ---------------------------------------------------------------------------
__global__ void k_degree(const int* __restrict__ dst, float* __restrict__ deg, int E) {
  int e = blockIdx.x * blockDim.x + threadIdx.x;
  if (e < E) atomicAdd(&deg[dst[e]], 1.0f);
}

__global__ void k_invsqrt(const float* __restrict__ deg, float* __restrict__ invs, int n) {
  int v = blockIdx.x * blockDim.x + threadIdx.x;
  if (v < n) {
    float d = deg[v];
    invs[v] = (d > 0.0f) ? rsqrtf(d) : 0.0f;
  }
}

// ---------------------------------------------------------------------------
// Edge scatter: one wave32 per edge. agg[v, dir*256 + c] += enorm * x[s,c]*rel[t,c]
// Lanes sweep channels in coalesced 32-float groups.
// ---------------------------------------------------------------------------
__global__ void __launch_bounds__(256)
k_scatter(const float* __restrict__ x,
          const float* __restrict__ rel_repr,
          const float* __restrict__ invs,
          const int* __restrict__ src, const int* __restrict__ dst,
          const int* __restrict__ etype, const int* __restrict__ edir,
          float* __restrict__ agg, int E) {
  int wave = (int)((blockIdx.x * (unsigned)blockDim.x + threadIdx.x) >> 5);
  int lane = threadIdx.x & 31;
  if (wave >= E) return;
  int s = src[wave];
  int d = dst[wave];
  int t = etype[wave];
  int dir = edir[wave] & 1;
  float en = invs[s] * invs[d];
  const float* xs = x + (size_t)s * IN_CH;
  const float* rr = rel_repr + (size_t)t * IN_CH;
  float* ap = agg + (size_t)d * (2 * IN_CH) + (size_t)dir * IN_CH;
#pragma unroll
  for (int it = 0; it < IN_CH / 32; ++it) {
    int c = it * 32 + lane;
    atomicAdd(&ap[c], xs[c] * rr[c] * en);
  }
}

// ---------------------------------------------------------------------------
// Pack W = [w0; w1; w2] (K=768, N=256) transposed + converted to f16:
// Bt[n*768 + k] = (half) w[d, k%256, n],  d = k/256.
// N-major rows let each WMMA B-fragment lane load 16 contiguous halfs.
// ---------------------------------------------------------------------------
__global__ void k_packB(const float* __restrict__ w, _Float16* __restrict__ Bt) {
  int idx = blockIdx.x * blockDim.x + threadIdx.x;
  if (idx >= KDIM * OUT_CH) return;
  int n = idx / KDIM;
  int k = idx % KDIM;
  int d = k / IN_CH;
  int kk = k % IN_CH;
  Bt[idx] = (_Float16)w[(size_t)d * IN_CH * OUT_CH + (size_t)kk * OUT_CH + n];
}

// ---------------------------------------------------------------------------
// WMMA GEMM: h_pre[V,256] = A[V,768] @ W[768,256], fp32 accumulate.
//   Phase 1 (k in [0,512)):   A = agg (scatter accumulators, fp32 -> f16)
//   Phase 2 (k in [512,768)): A = x * loop_rel (self-loop compose)
// Splitting the K loop at the uniform 512 boundary keeps the hot loops
// branch-free so the compiler can software-pipeline loads against WMMA.
// Block = 256 threads = 8 waves; waves tiled 2(M) x 4(N); wave tile 16x64.
// Epilogue: /3 + bias, store fp32 into d_out.
// A-fragment layout (f16 16x32): lanes 0-15 row M=lane, K in {0..15};
// lanes 16-31 same rows, K in {16..31}.  B (32x16): lane n=lane&15,
// K = (lane<16 ? 0 : 16) + i for half i of the v16h.
// ---------------------------------------------------------------------------
__global__ void __launch_bounds__(256)
k_gemm(const float* __restrict__ agg,
       const float* __restrict__ x,
       const float* __restrict__ loop_rel,
       const _Float16* __restrict__ Bt,
       const float* __restrict__ bias,
       float* __restrict__ hout) {
  const int lane = threadIdx.x & 31;
  const int wave = threadIdx.x >> 5;
  const int wm = wave >> 2;          // 0..1
  const int wn = wave & 3;           // 0..3
  const int row_base = blockIdx.x * 32 + wm * 16;
  const int col_base = wn * 64;

  const int m    = row_base + (lane & 15);
  const int kbA  = (lane < 16) ? 0 : 16;
  const int kbB  = (lane < 16) ? 0 : 16;
  const int ncol = lane & 15;

  v8f acc[4] = {};

  const float* aggRow = agg + (size_t)m * (2 * IN_CH);
  const float* xRow   = x   + (size_t)m * IN_CH;

  // -------- Phase 1: K = [0, 512), A fragments from fp32 accumulators ------
  for (int kk = 0; kk < 2 * IN_CH; kk += 32) {
    v16h a;
#pragma unroll
    for (int hr = 0; hr < 2; ++hr) {
      const int k0 = kk + hr * 8 + kbA;
      const float4* p = reinterpret_cast<const float4*>(aggRow + k0);
      const float4 f0 = p[0];
      const float4 f1 = p[1];
      a[hr * 8 + 0] = (_Float16)f0.x; a[hr * 8 + 1] = (_Float16)f0.y;
      a[hr * 8 + 2] = (_Float16)f0.z; a[hr * 8 + 3] = (_Float16)f0.w;
      a[hr * 8 + 4] = (_Float16)f1.x; a[hr * 8 + 5] = (_Float16)f1.y;
      a[hr * 8 + 6] = (_Float16)f1.z; a[hr * 8 + 7] = (_Float16)f1.w;
    }
#pragma unroll
    for (int f = 0; f < 4; ++f) {
      const int n = col_base + f * 16 + ncol;
      const v16h b = *reinterpret_cast<const v16h*>(Bt + (size_t)n * KDIM + kk + kbB);
      acc[f] = __builtin_amdgcn_wmma_f32_16x16x32_f16(
          false, a, false, b, (short)0, acc[f], false, false);
    }
  }

  // -------- Phase 2: K = [512, 768), A fragments = x * loop_rel ------------
  for (int kk = 2 * IN_CH; kk < KDIM; kk += 32) {
    v16h a;
#pragma unroll
    for (int hr = 0; hr < 2; ++hr) {
      const int kx = kk - 2 * IN_CH + hr * 8 + kbA;
      const float4* px = reinterpret_cast<const float4*>(xRow + kx);
      const float4* pl = reinterpret_cast<const float4*>(loop_rel + kx);
      const float4 xa = px[0], xb = px[1];
      const float4 la = pl[0], lb = pl[1];
      a[hr * 8 + 0] = (_Float16)(xa.x * la.x); a[hr * 8 + 1] = (_Float16)(xa.y * la.y);
      a[hr * 8 + 2] = (_Float16)(xa.z * la.z); a[hr * 8 + 3] = (_Float16)(xa.w * la.w);
      a[hr * 8 + 4] = (_Float16)(xb.x * lb.x); a[hr * 8 + 5] = (_Float16)(xb.y * lb.y);
      a[hr * 8 + 6] = (_Float16)(xb.z * lb.z); a[hr * 8 + 7] = (_Float16)(xb.w * lb.w);
    }
#pragma unroll
    for (int f = 0; f < 4; ++f) {
      const int n = col_base + f * 16 + ncol;
      const v16h b = *reinterpret_cast<const v16h*>(Bt + (size_t)n * KDIM + kk + kbB);
      acc[f] = __builtin_amdgcn_wmma_f32_16x16x32_f16(
          false, a, false, b, (short)0, acc[f], false, false);
    }
  }

  // ---- epilogue: h_pre = acc/3 + bias ----
  const float scale = 1.0f / 3.0f;
  const int rshift = (lane < 16) ? 0 : 8;
#pragma unroll
  for (int f = 0; f < 4; ++f) {
    const int n = col_base + f * 16 + ncol;
    const float bv = bias[n];
#pragma unroll
    for (int r = 0; r < 8; ++r) {
      const int row = row_base + r + rshift;
      hout[(size_t)row * OUT_CH + n] = acc[f][r] * scale + bv;
    }
  }
}

// ---------------------------------------------------------------------------
// BatchNorm statistics: per-channel sum and sum-of-squares over V rows.
// ---------------------------------------------------------------------------
__global__ void __launch_bounds__(256)
k_bnstats(const float* __restrict__ h, float* __restrict__ sum, float* __restrict__ sumsq) {
  int c = threadIdx.x;                       // 0..255 = channel
  float s = 0.0f, q = 0.0f;
  for (int r = blockIdx.x; r < V_NODES; r += gridDim.x) {
    float v = h[(size_t)r * OUT_CH + c];
    s += v;
    q += v * v;
  }
  atomicAdd(&sum[c], s);
  atomicAdd(&sumsq[c], q);
}

__global__ void k_bnfinal(const float* __restrict__ sum, const float* __restrict__ sumsq,
                          float* __restrict__ mean, float* __restrict__ istd) {
  int c = threadIdx.x;
  float m = sum[c] * (1.0f / (float)V_NODES);
  float var = sumsq[c] * (1.0f / (float)V_NODES) - m * m;
  mean[c] = m;
  istd[c] = rsqrtf(var + BN_EPS);
}

__global__ void k_bnnorm(float* __restrict__ h,
                         const float* __restrict__ mean, const float* __restrict__ istd) {
  long long i = (long long)blockIdx.x * blockDim.x + threadIdx.x;
  const long long n = (long long)V_NODES * OUT_CH;
  long long stride = (long long)gridDim.x * blockDim.x;
  for (; i < n; i += stride) {
    int c = (int)(i & (OUT_CH - 1));
    h[i] = (h[i] - mean[c]) * istd[c];
  }
}

// ---------------------------------------------------------------------------
// rel_out[i] = dot(rel_all[i], w_rel[:, 255]); rel_all = [rel_repr; loop_rel]
// ---------------------------------------------------------------------------
__global__ void k_relout(const float* __restrict__ rel_repr,
                         const float* __restrict__ loop_rel,
                         const float* __restrict__ w_rel,
                         float* __restrict__ out) {
  int i = blockIdx.x * blockDim.x + threadIdx.x;
  if (i >= NREL2 + 1) return;
  const float* row = (i < NREL2) ? (rel_repr + (size_t)i * IN_CH) : loop_rel;
  float s = 0.0f;
  for (int k = 0; k < IN_CH; ++k)
    s += row[k] * w_rel[(size_t)k * OUT_CH + (OUT_CH - 1)];
  out[(size_t)V_NODES * OUT_CH + i] = s;
}

// ---------------------------------------------------------------------------
extern "C" void kernel_launch(void* const* d_in, const int* in_sizes, int n_in,
                              void* d_out, int out_size, void* d_ws, size_t ws_size,
                              hipStream_t stream) {
  const float* x        = (const float*)d_in[0];
  const float* rel_repr = (const float*)d_in[1];
  const float* w        = (const float*)d_in[2];   // [3,256,256]
  const float* w_rel    = (const float*)d_in[3];   // [256,256]
  const float* loop_rel = (const float*)d_in[4];   // [1,256]
  const float* bias     = (const float*)d_in[5];   // [256]
  const int*   src      = (const int*)d_in[6];
  const int*   dst      = (const int*)d_in[7];
  const int*   etype    = (const int*)d_in[8];
  const int*   edir     = (const int*)d_in[9];
  float* out = (float*)d_out;

  // --- carve workspace (keep every sub-buffer 256B aligned) ---
  char* ws = (char*)d_ws;
  size_t off = 0;
  auto take = [&](size_t bytes) -> void* {
    void* p = (void*)(ws + off);
    off += (bytes + 255) & ~(size_t)255;
    return p;
  };
  float*     deg      = (float*)take((size_t)V_NODES * 4);
  float*     invs     = (float*)take((size_t)V_NODES * 4);
  float*     agg      = (float*)take((size_t)V_NODES * 2 * IN_CH * 4);   // ~205 MB
  _Float16*  Bt       = (_Float16*)take((size_t)KDIM * OUT_CH * 2);
  float*     bn_sum   = (float*)take(256 * 4);
  float*     bn_sumsq = (float*)take(256 * 4);   // contiguous with bn_sum
  float*     bn_mean  = (float*)take(256 * 4);
  float*     bn_istd  = (float*)take(256 * 4);
  (void)ws_size; (void)in_sizes; (void)n_in; (void)out_size;

  // 1) zero accumulators
  k_zero<<<2048, 256, 0, stream>>>(agg, (long long)V_NODES * 2 * IN_CH);
  k_zero<<<(V_NODES + 255) / 256, 256, 0, stream>>>(deg, (long long)V_NODES);
  k_zero<<<2, 256, 0, stream>>>(bn_sum, 512);   // bn_sum + bn_sumsq (contiguous)

  // 2) degrees + symmetric norm
  k_degree<<<(E_EDGES + 255) / 256, 256, 0, stream>>>(dst, deg, E_EDGES);
  k_invsqrt<<<(V_NODES + 255) / 256, 256, 0, stream>>>(deg, invs, V_NODES);

  // 3) composed-message scatter (one wave32 per edge, 8 edges/block)
  k_scatter<<<E_EDGES / 8, 256, 0, stream>>>(x, rel_repr, invs, src, dst,
                                             etype, edir, agg, E_EDGES);

  // 4) pack fused weight matrix to f16 (N-major for B fragments)
  k_packB<<<(KDIM * OUT_CH + 255) / 256, 256, 0, stream>>>(w, Bt);

  // 5) WMMA GEMM + epilogue (h_pre -> d_out)
  k_gemm<<<V_NODES / 32, 256, 0, stream>>>(agg, x, loop_rel, Bt, bias, out);

  // 6) BatchNorm (training-mode batch stats)
  k_bnstats<<<512, 256, 0, stream>>>(out, bn_sum, bn_sumsq);
  k_bnfinal<<<1, 256, 0, stream>>>(bn_sum, bn_sumsq, bn_mean, bn_istd);
  k_bnnorm<<<2048, 256, 0, stream>>>(out, bn_mean, bn_istd);

  // 7) relation output (475 dot products)
  k_relout<<<2, 256, 0, stream>>>(rel_repr, loop_rel, w_rel, out);
}
